// LSTM_55508157334228
// MI455X (gfx1250) — matrix operand order repaired and checked by reading, not compiled, not run on hardware
//
#include <hip/hip_runtime.h>
#include <hip/hip_bf16.h>

// Problem constants (fixed by the reference).
#define BB 2048
#define TT 512
#define II 64
#define HH 32

typedef __attribute__((ext_vector_type(16))) __bf16 v16bf;
typedef __attribute__((ext_vector_type(8)))  __bf16 v8bf;
typedef __attribute__((ext_vector_type(8)))  float  v8f;

#if defined(__AMDGCN__) && __has_builtin(__builtin_amdgcn_tensor_load_to_lds) && \
    __has_builtin(__builtin_amdgcn_s_wait_tensorcnt)
#define HAVE_TDM 1
#else
#define HAVE_TDM 0
#endif

#if HAVE_TDM
typedef unsigned int u32x4 __attribute__((ext_vector_type(4)));
typedef int          i32x4 __attribute__((ext_vector_type(4)));
typedef int          i32x8 __attribute__((ext_vector_type(8)));

// 2D TDM load: tile = 16 rows x 64 f32, row stride TT*II elements.
// gptr = global address of tile start, lds = byte address in LDS.
__device__ __forceinline__ void tdm_load_x(unsigned int lds, const float* gptr) {
  unsigned long long ga = (unsigned long long)(uintptr_t)gptr;
  u32x4 g0 = { 1u,                                  // count=1 valid descriptor
               lds,                                 // lds_addr (bytes)
               (unsigned int)ga,                    // global_addr[31:0]
               ((unsigned int)(ga >> 32) & 0x01FFFFFFu) | 0x80000000u }; // [56:32] | type=2
  i32x8 g1 = { (int)0x20000,        // data_size=2 (4B), no multicast/pad/iterate
               (int)(64u << 16),    // tensor_dim0 = 64 (low16 in [63:48])
               (int)(16u << 16),    // tensor_dim1 = 16 (low16 in [95:80])
               (int)(64u << 16),    // tile_dim0 = 64   ([127:112])
               (int)16,             // tile_dim1 = 16, tile_dim2 = 0
               (int)(TT * II),      // tensor_dim0_stride low32 = 32768
               0, 0 };
  i32x4 gz = { 0, 0, 0, 0 };
#if __clang_major__ >= 23
  i32x8 gz8 = { 0, 0, 0, 0, 0, 0, 0, 0 };
  __builtin_amdgcn_tensor_load_to_lds(g0, g1, gz, gz, gz8, 0);
#else
  __builtin_amdgcn_tensor_load_to_lds(g0, g1, gz, gz, 0);
#endif
}
#endif  // HAVE_TDM

__device__ __forceinline__ float tanh_fast(float x) {
#if __has_builtin(__builtin_amdgcn_tanhf)
  return __builtin_amdgcn_tanhf(x);          // v_tanh_f32 on gfx1250
#else
  float e = __expf(x + x);
  return 1.0f - 2.0f / (e + 1.0f);
#endif
}
__device__ __forceinline__ float sigm_fast(float x) {
  return fmaf(tanh_fast(0.5f * x), 0.5f, 0.5f);
}

// Assemble a 16-half WMMA operand from two contiguous 8-half LDS runs.
__device__ __forceinline__ v16bf ld_op16(const __bf16* p) {
  const v8bf lo = *(const v8bf*)p;
  const v8bf hi = *(const v8bf*)(p + 16);
  return __builtin_shufflevector(lo, hi, 0, 1, 2, 3, 4, 5, 6, 7,
                                 8, 9, 10, 11, 12, 13, 14, 15);
}

// One wave (32 lanes) owns a 16-row batch tile for all 512 timesteps.
// g = [16 x 128] = 8 WMMA column tiles; W_ih streamed from LDS (off the
// recurrence critical path), W_hh resident in VGPRs (on it).
__global__ __launch_bounds__(32, 1)
void lstm_fused(const float* __restrict__ x,   const float* __restrict__ Wih,
                const float* __restrict__ Whh, const float* __restrict__ bih,
                const float* __restrict__ bhh, const float* __restrict__ Wfc,
                const float* __restrict__ bfc, float* __restrict__ out)
{
  const int lane = threadIdx.x & 31;
  const int nsub = lane & 15;      // N (B/C/D) or M (A) sub-index
  const int hi   = lane >> 4;      // lane half
  const int kbA  = hi * 8;         // A-operand K base (interleaved pattern)
  const int row0 = blockIdx.x * 16;

  __shared__ __bf16 WihL[128 * II];  // W_ih row-major [col][k], bf16 (16 KB)
  __shared__ __bf16 hL[16 * HH];     // h transpose buffer, row-major [m][n]
#if HAVE_TDM
  __shared__ float  xT[4 * 16 * II]; // 4-deep ring of x tiles [16][64] (16 KB)
#endif

  // ---------- Stage W_ih into LDS as bf16 ----------
  for (int i = lane; i < 128 * II; i += 32) WihL[i] = (__bf16)Wih[i];

  // ---------- W_hh into VGPRs (B-operand layout: lane=col n, K interleaved) ----------
  v16bf WhhB[8];
  #pragma unroll
  for (int t8 = 0; t8 < 8; ++t8) {
    const float* wp = Whh + (size_t)(16 * t8 + nsub) * HH + 8 * hi;
    v16bf w;
    #pragma unroll
    for (int j = 0; j < 16; ++j) w[j] = (__bf16)wp[j + (j & 8)];
    WhhB[t8] = w;
  }
  float bias[8];
  #pragma unroll
  for (int t8 = 0; t8 < 8; ++t8)
    bias[t8] = bih[16 * t8 + nsub] + bhh[16 * t8 + nsub];

  __syncthreads();   // WihL ready

  // ---------- Recurrent state ----------
  v8f c0 = {}; v8f c1 = {};
  v16bf hA;
  #pragma unroll
  for (int j = 0; j < 16; ++j) hA[j] = (__bf16)0.0f;

  const float* gx0 = x + (size_t)row0 * TT * II;   // block's x base (uniform)

#if HAVE_TDM
  const unsigned int xT_base = (unsigned int)(uintptr_t)&xT[0];
  #pragma unroll
  for (int s = 0; s < 4; ++s)
    tdm_load_x(xT_base + (unsigned int)s * 16 * II * 4, gx0 + (size_t)s * II);
#else
  const float* xrow = gx0 + (size_t)nsub * TT * II;
  float4 xv[8], xn[8];
  auto loadx = [&](float4* dst, int t) {
    const float* p = xrow + (size_t)t * II;
    #pragma unroll
    for (int cc = 0; cc < 2; ++cc)
      #pragma unroll
      for (int g2 = 0; g2 < 2; ++g2)
        #pragma unroll
        for (int q = 0; q < 2; ++q)
          dst[cc * 4 + g2 * 2 + q] =
              *(const float4*)(p + cc * 32 + g2 * 16 + kbA + q * 4);
  };
  loadx(xv, 0);
#endif

  for (int t = 0; t < TT; ++t) {
    // ---- x_t into two A operands (K 0..31, 32..63), bf16 ----
    float4 xq[8];
#if HAVE_TDM
    __builtin_amdgcn_s_wait_tensorcnt(3);   // oldest ring slot complete
    asm volatile("" ::: "memory");          // TDM wrote LDS behind compiler's back
    {
      const float* xs = &xT[(t & 3) * (16 * II)] + nsub * II;
      #pragma unroll
      for (int cc = 0; cc < 2; ++cc)
        #pragma unroll
        for (int g2 = 0; g2 < 2; ++g2)
          #pragma unroll
          for (int q = 0; q < 2; ++q)
            xq[cc * 4 + g2 * 2 + q] =
                *(const float4*)(xs + cc * 32 + g2 * 16 + kbA + q * 4);
    }
#else
    if (t + 1 < TT) loadx(xn, t + 1);
    #pragma unroll
    for (int q = 0; q < 8; ++q) xq[q] = xv[q];
#endif
    v16bf xa0, xa1;
    #pragma unroll
    for (int j = 0; j < 16; ++j) {
      const int idx  = (j >> 3) * 2 + ((j & 7) >> 2);
      const int comp = j & 3;
      const float4 f0 = xq[idx];
      const float4 f1 = xq[4 + idx];
      const float a0 = (comp == 0) ? f0.x : (comp == 1) ? f0.y : (comp == 2) ? f0.z : f0.w;
      const float a1 = (comp == 0) ? f1.x : (comp == 1) ? f1.y : (comp == 2) ? f1.z : f1.w;
      xa0[j] = (__bf16)a0;
      xa1[j] = (__bf16)a1;
    }
#if HAVE_TDM
    if (t + 4 < TT)   // x_t consumed into VGPRs -> slot is reusable
      tdm_load_x(xT_base + (unsigned int)(t & 3) * 16 * II * 4,
                 gx0 + (size_t)(t + 4) * II);
#endif

    // ---- two gate groups: tiles {0,2,4,6} -> hidden 0..15, {1,3,5,7} -> 16..31 ----
    #pragma unroll
    for (int grp = 0; grp < 2; ++grp) {
      v8f* cs = grp ? &c1 : &c0;
      v8f a[4];
      #pragma unroll
      for (int u = 0; u < 4; ++u) {
        const int t8 = 2 * u + grp;
        const __bf16* wb = &WihL[(size_t)(16 * t8 + nsub) * II + 8 * hi];
        const v16bf b0 = ld_op16(wb);
        const v16bf b1 = ld_op16(wb + 32);
        v8f acc;
        #pragma unroll
        for (int j = 0; j < 8; ++j) acc[j] = bias[t8];
        acc = __builtin_amdgcn_wmma_f32_16x16x32_bf16(false, xa0, false, b0,
                                                      (short)0, acc, false, false);
        acc = __builtin_amdgcn_wmma_f32_16x16x32_bf16(false, xa1, false, b1,
                                                      (short)0, acc, false, false);
        acc = __builtin_amdgcn_wmma_f32_16x16x32_bf16(false, hA,  false, WhhB[t8],
                                                      (short)0, acc, false, false);
        a[u] = acc;
      }
      #pragma unroll
      for (int j = 0; j < 8; ++j) {
        const float iv = sigm_fast(a[0][j]);
        const float fv = sigm_fast(a[1][j]);
        const float gv = tanh_fast(a[2][j]);
        const float ov = sigm_fast(a[3][j]);
        (*cs)[j] = fmaf(fv, (*cs)[j], iv * gv);
        const float hv = ov * tanh_fast((*cs)[j]);
        hL[(j + 8 * hi) * HH + 16 * grp + nsub] = (__bf16)hv;
      }
    }

    __syncthreads();   // D-layout stores -> A-layout loads
    hA = ld_op16(&hL[nsub * HH + kbA]);      // note: runs are 8 apart => p, p+8?
    // A-operand runs are {kbA..kbA+7} and {kbA+16..kbA+23}: load explicitly.
    {
      const __bf16* hp = &hL[nsub * HH + kbA];
      const v8bf lo  = *(const v8bf*)hp;
      const v8bf hi8 = *(const v8bf*)(hp + 16);
      hA = __builtin_shufflevector(lo, hi8, 0, 1, 2, 3, 4, 5, 6, 7,
                                   8, 9, 10, 11, 12, 13, 14, 15);
    }
    __syncthreads();   // protect hL against next iteration's stores

#if !HAVE_TDM
    #pragma unroll
    for (int q = 0; q < 8; ++q) xv[q] = xn[q];
#endif
  }

  // ---------- Final fc from hL (bf16): out[b] = h . W_fc + b_fc ----------
  if (lane < 16) {
    float s = bfc[0];
    #pragma unroll
    for (int n = 0; n < HH; ++n)
      s = fmaf((float)hL[lane * HH + n], Wfc[n], s);
    out[row0 + lane] = s;
  }
}

extern "C" void kernel_launch(void* const* d_in, const int* in_sizes, int n_in,
                              void* d_out, int out_size, void* d_ws, size_t ws_size,
                              hipStream_t stream) {
  const float* x    = (const float*)d_in[0];
  const float* Wih  = (const float*)d_in[1];
  const float* Whh  = (const float*)d_in[2];
  const float* bih  = (const float*)d_in[3];
  const float* bhh  = (const float*)d_in[4];
  const float* Wfc  = (const float*)d_in[5];
  const float* bfc  = (const float*)d_in[6];
  float* out = (float*)d_out;

  dim3 grid(BB / 16);   // 128 waves, one 16-row batch tile each
  dim3 block(32);       // one wave32 per block
  lstm_fused<<<grid, block, 0, stream>>>(x, Wih, Whh, bih, bhh, Wfc, bfc, out);
}
